// Head_73169062855352
// MI455X (gfx1250) — matrix operand order repaired and checked by reading, not compiled, not run on hardware
//
#include <hip/hip_runtime.h>

// MI455X / gfx1250: wave32, WMMA bf16 16x16x32, fp32 accumulate.
typedef __attribute__((ext_vector_type(16))) __bf16 v16bf;
typedef __attribute__((ext_vector_type(8)))  __bf16 v8bf;
typedef __attribute__((ext_vector_type(2)))  __bf16 v2bf;
typedef __attribute__((ext_vector_type(8)))  float  v8f;
typedef __attribute__((ext_vector_type(4)))  unsigned v4u;

#define TT 4096
#define EE 1024
#define HH 64
#define BBATCH 4
#define NROW (BBATCH * TT)

// Pack two f32 -> two bf16 in one dword.
// Native v_cvt_pk if the toolchain exposes it; else round-half-up + v_perm_b32
// (byte-select the two high halves in a single VALU op).
__device__ __forceinline__ unsigned pk2bf(float f0, float f1) {
#if __has_builtin(__builtin_amdgcn_cvt_pk_bf16_f32)
  v2bf p = __builtin_amdgcn_cvt_pk_bf16_f32(f0, f1);
  return __builtin_bit_cast(unsigned, p);
#else
  unsigned u0 = __builtin_bit_cast(unsigned, f0) + 0x8000u;
  unsigned u1 = __builtin_bit_cast(unsigned, f1) + 0x8000u;
  // {S0,S1} byte pool: sel 0-3 -> S1 bytes, 4-7 -> S0 bytes.
  // dst = [u0.b2, u0.b3, u1.b2, u1.b3] = {bf(f0), bf(f1)}
  return __builtin_amdgcn_perm(u1, u0, 0x07060302u);
#endif
}

__device__ __forceinline__ __bf16 f2bf(float f) {
  unsigned u = (__builtin_bit_cast(unsigned, f) + 0x8000u) >> 16;
  unsigned short s = (unsigned short)u;
  return __builtin_bit_cast(__bf16, s);
}

__device__ __forceinline__ v8bf cvt_v8(v8f x) {
  v4u r;
#pragma unroll
  for (int e = 0; e < 4; ++e) r[e] = pk2bf(x[2 * e], x[2 * e + 1]);
  return __builtin_bit_cast(v8bf, r);
}

__device__ __forceinline__ v16bf cat8(v8bf lo, v8bf hi) {
  return __builtin_shufflevector(lo, hi, 0,1,2,3,4,5,6,7,8,9,10,11,12,13,14,15);
}

__device__ __forceinline__ v8f wmma_bf16(v16bf a, v16bf b, v8f c) {
  return __builtin_amdgcn_wmma_f32_16x16x32_bf16(false, a, false, b, (short)0, c,
                                                 false, false);
}

// ---------------------------------------------------------------------------
// Kernel 1: fused K/Q/V projections. grid = NROW/16, block = 128 (4 waves).
// Each wave: one 16(t) x 16(h) tile for ALL THREE projections; the x A-frag
// is loaded+converted once per k-step and reused for 3 WMMAs. All 8 fragment
// loads are issued before any conversion to expose memory-level parallelism.
//   Kb = (x @ Wk^T) * (1/sqrt(E)) row-major [NROW,H] bf16
//   Qb =  x @ Wq^T               row-major [NROW,H] bf16
//   Vt = (x @ Wv^T)^T per batch  [B][H][T] bf16 (contiguous in t)
// ---------------------------------------------------------------------------
__global__ __launch_bounds__(128) void hk_proj(
    const float* __restrict__ x,
    const float* __restrict__ Wk, const float* __restrict__ Wq,
    const float* __restrict__ Wv,
    __bf16* __restrict__ Kb, __bf16* __restrict__ Qb, __bf16* __restrict__ Vt)
{
  const int row0 = blockIdx.x << 4;        // global row in flattened [B*T]
  const int lane = threadIdx.x & 31;
  const int wv   = threadIdx.x >> 5;
  const int hb   = wv << 4;                // h-tile base
  const int nl   = lane & 15;
  const int hv   = lane >> 4;              // lane half
  const int kb8  = hv << 4;                // A-frag K split (contiguous)
  const int kb16 = hv << 4;                // B-frag K split

  const float* arow = x  + (size_t)(row0 + nl) * EE;  // A: lane holds row M=nl
  const float* bk   = Wk + (size_t)(hb + nl) * EE;    // B: lane holds col N=nl
  const float* bq   = Wq + (size_t)(hb + nl) * EE;
  const float* bv   = Wv + (size_t)(hb + nl) * EE;

  v8f ack = {0.f,0.f,0.f,0.f,0.f,0.f,0.f,0.f};
  v8f acq = ack, acv = ack;

  for (int eb = 0; eb < EE; eb += 32) {
    // issue every load first (2 for A, 2 per weight matrix)
    v8f la0 = *(const v8f*)(arow + eb + kb8);
    v8f la1 = *(const v8f*)(arow + eb + kb8 + 8);
    v8f lk0 = *(const v8f*)(bk + eb + kb16);
    v8f lk1 = *(const v8f*)(bk + eb + kb16 + 8);
    v8f lq0 = *(const v8f*)(bq + eb + kb16);
    v8f lq1 = *(const v8f*)(bq + eb + kb16 + 8);
    v8f lv0 = *(const v8f*)(bv + eb + kb16);
    v8f lv1 = *(const v8f*)(bv + eb + kb16 + 8);

    // A 16x32: elems 0-15 -> K=kb8+e
    v16bf a = cat8(cvt_v8(la0), cvt_v8(la1));
    // B 32x16: elems 0-15 -> K=kb16+e
    ack = wmma_bf16(a, cat8(cvt_v8(lk0), cvt_v8(lk1)), ack);
    acq = wmma_bf16(a, cat8(cvt_v8(lq0), cvt_v8(lq1)), acq);
    acv = wmma_bf16(a, cat8(cvt_v8(lv0), cvt_v8(lv1)), acv);
  }

  // C layout: lane holds column N=nl; rows M = hv*8 + r in acc[r]
#pragma unroll
  for (int r = 0; r < 8; ++r) {
    const size_t ro = (size_t)(row0 + (hv << 3) + r) * HH + hb + nl;
    Kb[ro] = f2bf(ack[r] * 0.03125f);    // fold 1/sqrt(1024) into K
    Qb[ro] = f2bf(acq[r]);
  }
  const int bb = row0 >> 12;             // T == 4096
  const int t0 = row0 & (TT - 1);
  v8bf pv;
#pragma unroll
  for (int r = 0; r < 8; ++r) pv[r] = f2bf(acv[r]);
  *(v8bf*)(Vt + (size_t)(bb * HH + hb + nl) * TT + t0 + (hv << 3)) = pv;
}

// ---------------------------------------------------------------------------
// Kernel 2: per-column softmax stats (softmax over i, for each column j).
// grid = NROW/64, block = 128; each wave owns 16 columns, streams i-tiles
// from the diagonal tile to T. Writes m_j and 1/Z_j.
// ---------------------------------------------------------------------------
__global__ __launch_bounds__(128) void hk_stats(
    const __bf16* __restrict__ Kb, const __bf16* __restrict__ Qb,
    float* __restrict__ Mj, float* __restrict__ Rz)
{
  const int lane = threadIdx.x & 31;
  const int wv   = threadIdx.x >> 5;
  const int colg = (blockIdx.x << 6) + (wv << 4);  // global column row index
  const int bb   = colg >> 12;
  const int j0   = colg & (TT - 1);
  const int nl   = lane & 15;
  const int hv   = lane >> 4;
  const int kb8  = hv << 4;
  const int kb16 = hv << 4;

  const __bf16* Kbase = Kb + (size_t)bb * TT * HH;
  // B operand = Q^T: lane holds column N=nl (query j0+nl), K = h contiguous
  const __bf16* Qrow = Qb + (size_t)(colg + nl) * HH;
  const v16bf qb0 = *(const v16bf*)(Qrow + kb16);        // h in [0,32)
  const v16bf qb1 = *(const v16bf*)(Qrow + 32 + kb16);   // h in [32,64)

  float m_run = -3.0e38f;
  float Z     = 0.0f;
  const int jcol = j0 + nl;

  for (int i0 = j0; i0 < TT; i0 += 16) {
    const __bf16* Arow = Kbase + (size_t)(i0 + nl) * HH; // A: row M=nl (key i)
    __builtin_prefetch(Arow + 16 * HH, 0, 1);            // next i-tile
    v16bf a0 = cat8(*(const v8bf*)(Arow + kb8),
                    *(const v8bf*)(Arow + kb8 + 8));
    v16bf a1 = cat8(*(const v8bf*)(Arow + 32 + kb8),
                    *(const v8bf*)(Arow + 32 + kb8 + 8));
    v8f c = {0.f,0.f,0.f,0.f,0.f,0.f,0.f,0.f};
    c = wmma_bf16(a0, qb0, c);
    c = wmma_bf16(a1, qb1, c);

    if (i0 == j0) {                      // diagonal tile: mask i < j
#pragma unroll
      for (int r = 0; r < 8; ++r) {
        const int i = i0 + (hv << 3) + r;
        if (i < jcol) c[r] = -3.0e38f;
      }
    }
    // online softmax update per column (lane pair l, l+16 holds same column)
    float tm = c[0];
#pragma unroll
    for (int r = 1; r < 8; ++r) tm = fmaxf(tm, c[r]);
    tm = fmaxf(tm, __shfl_xor(tm, 16, 32));
    const float m_new = fmaxf(m_run, tm);
    const float fre   = __expf(m_run - m_new);
    float sloc = 0.f;
#pragma unroll
    for (int r = 0; r < 8; ++r) sloc += __expf(c[r] - m_new);
    sloc += __shfl_xor(sloc, 16, 32);
    Z = Z * fre + sloc;
    m_run = m_new;
  }
  if (hv == 0) {
    Mj[colg + nl] = m_run;
    Rz[colg + nl] = 1.0f / Z;            // diagonal always valid => Z > 0
  }
}

// ---------------------------------------------------------------------------
// Kernel 3: out[i,h] = sum_{j<=i} exp(s[i,j]-m_j)/Z_j * v[j,h].
// grid = NROW/64, block = 128; each wave owns 16 rows x all 64 h.
// Per 32-wide j step: 4 WMMAs for S, exp-weights -> per-wave LDS tile,
// transpose read as A fragment, 4 WMMAs against V^T.
// ---------------------------------------------------------------------------
__global__ __launch_bounds__(128) void hk_attnout(
    const __bf16* __restrict__ Kb, const __bf16* __restrict__ Qb,
    const __bf16* __restrict__ Vt,
    const float* __restrict__ Mj, const float* __restrict__ Rz,
    float* __restrict__ out)
{
  __shared__ __bf16 wtile[4 * 16 * 40];  // per-wave 16x32 W tile, stride 40
  const int lane = threadIdx.x & 31;
  const int wv   = threadIdx.x >> 5;
  const int rowg = (blockIdx.x << 6) + (wv << 4);
  const int bb   = rowg >> 12;
  const int i0   = rowg & (TT - 1);
  const int nl   = lane & 15;
  const int hv   = lane >> 4;
  const int kb8  = hv << 4;
  const int kb16 = hv << 4;

  const __bf16* Kbase = Kb + (size_t)bb * TT * HH;
  const __bf16* Qbase = Qb + (size_t)bb * TT * HH;
  const __bf16* Vbase = Vt + (size_t)bb * HH * TT;
  const float*  Mb    = Mj + (size_t)bb * TT;
  const float*  Rb    = Rz + (size_t)bb * TT;

  // K rows for this wave's 16 output rows: invariant over the j loop
  const __bf16* Arow = Kbase + (size_t)(i0 + nl) * HH;
  const v16bf ka0 = cat8(*(const v8bf*)(Arow + kb8),
                         *(const v8bf*)(Arow + kb8 + 8));
  const v16bf ka1 = cat8(*(const v8bf*)(Arow + 32 + kb8),
                         *(const v8bf*)(Arow + 32 + kb8 + 8));

  v8f acc[4] = {};                       // 16 rows x 64 h accumulators
  __bf16* wl = &wtile[wv * 16 * 40];

  for (int j0 = 0; j0 <= i0 + 15; j0 += 32) {   // max j touched = 4095
#pragma unroll
    for (int jt = 0; jt < 2; ++jt) {
      const int jc = j0 + (jt << 4);
      const __bf16* Qrow = Qbase + (size_t)(jc + nl) * HH;
      __builtin_prefetch(Qrow + 32 * HH, 0, 1);         // next j-step
      v16bf qb0 = *(const v16bf*)(Qrow + kb16);
      v16bf qb1 = *(const v16bf*)(Qrow + 32 + kb16);
      v8f c = {0.f,0.f,0.f,0.f,0.f,0.f,0.f,0.f};
      c = wmma_bf16(ka0, qb0, c);
      c = wmma_bf16(ka1, qb1, c);

      const int   j  = jc + nl;
      const float mj = Mb[j];
      const float rz = Rb[j];
#pragma unroll
      for (int r = 0; r < 8; ++r) {
        const int   i = i0 + (hv << 3) + r;
        const float w = (j <= i) ? __expf(c[r] - mj) * rz : 0.0f;
        wl[((hv << 3) + r) * 40 + (jt << 4) + nl] = f2bf(w);
      }
    }
    // same-wave LDS producer/consumer: wait for the b16 stores
    asm volatile("s_wait_dscnt 0" ::: "memory");

    // read W back as an A fragment (row M=nl, K=j contiguous, 16 per lane half)
    const __bf16* wrow = wl + nl * 40;
    v16bf wa = cat8(*(const v8bf*)(wrow + kb8),
                    *(const v8bf*)(wrow + kb8 + 8));
#pragma unroll
    for (int ht = 0; ht < 4; ++ht) {
      // B = V^T tile: lane holds column N = h, K = j contiguous
      const __bf16* vrow = Vbase + (size_t)((ht << 4) + nl) * TT + j0 + kb16;
      v16bf vb = *(const v16bf*)vrow;
      acc[ht] = wmma_bf16(wa, vb, acc[ht]);
    }
    asm volatile("s_wait_dscnt 0" ::: "memory");
  }

#pragma unroll
  for (int ht = 0; ht < 4; ++ht)
#pragma unroll
    for (int r = 0; r < 8; ++r)
      out[(size_t)(rowg + (hv << 3) + r) * HH + (ht << 4) + nl] = acc[ht][r];
}

// ---------------------------------------------------------------------------
extern "C" void kernel_launch(void* const* d_in, const int* in_sizes, int n_in,
                              void* d_out, int out_size, void* d_ws,
                              size_t ws_size, hipStream_t stream) {
  (void)in_sizes; (void)n_in; (void)out_size; (void)ws_size;
  const float* x  = (const float*)d_in[0];
  const float* Wk = (const float*)d_in[1];
  const float* Wq = (const float*)d_in[2];
  const float* Wv = (const float*)d_in[3];
  float* out = (float*)d_out;

  char* ws = (char*)d_ws;
  const size_t szKQ = (size_t)NROW * HH * sizeof(__bf16);   // 2 MB each
  __bf16* Kb = (__bf16*)(ws);
  __bf16* Qb = (__bf16*)(ws + szKQ);
  __bf16* Vt = (__bf16*)(ws + 2 * szKQ);
  float*  Mj = (float*)(ws + 3 * szKQ);
  float*  Rz = (float*)(ws + 3 * szKQ + (size_t)NROW * sizeof(float));

  hk_proj<<<dim3(NROW / 16), 128, 0, stream>>>(x, Wk, Wq, Wv, Kb, Qb, Vt);
  hk_stats<<<dim3(NROW / 64), 128, 0, stream>>>(Kb, Qb, Mj, Rz);
  hk_attnout<<<dim3(NROW / 64), 128, 0, stream>>>(Kb, Qb, Vt, Mj, Rz, out);
}
